// Mesh2GridGNN_58171037057097
// MI455X (gfx1250) — compile-verified, matrix operand order checked by reading, not run on hardware
//
#include <hip/hip_runtime.h>
#include <hip/hip_bf16.h>

#define N_MESH 10000
#define N_GRID 100000
#define N_EDGE 300000
#define DIM    256
#define HID    256

typedef __attribute__((ext_vector_type(16))) _Float16 v16h;
typedef __attribute__((ext_vector_type(8)))  float    v8f;

// ---------------------------------------------------------------------------
// 16-bit A/B fragment loader, CDNA5 16x16x32 layout (CDNA5 ISA):
//   lane L: row = L&15, hi = L>>4
//   VGPR i<4 : K = 2i   + 8*hi, 2i+1   + 8*hi
//   VGPR i>=4: K = 16+2(i-4)+8*hi, +1
// r points at (row, kbase) of a row-major matrix (LDS or global).
// B fragments use the transposed weight matrix so the same pattern applies.
// ---------------------------------------------------------------------------
template <typename P>
__device__ __forceinline__ v16h frag16(const P* r, int hi) {
  v16h a;
#pragma unroll
  for (int i = 0; i < 8; ++i) {
    int k0 = ((i < 4) ? (2 * i) : (16 + 2 * (i - 4))) + 8 * hi;
    a[2 * i]     = r[k0];
    a[2 * i + 1] = r[k0 + 1];
  }
  return a;
}

__device__ __forceinline__ v8f wmma_f16(v16h a, v16h b, v8f c) {
  return __builtin_amdgcn_wmma_f32_16x16x32_f16(
      /*neg_a=*/false, a, /*neg_b=*/false, b,
      /*c_mod=*/(short)0, c, /*reuse_a=*/false, /*reuse_b=*/false);
}

// ---------------------------------------------------------------------------
// Utility kernels
// ---------------------------------------------------------------------------
__global__ void zero_f32_kernel(float* __restrict__ p, int n) {
  for (int i = blockIdx.x * blockDim.x + threadIdx.x; i < n;
       i += gridDim.x * blockDim.x)
    p[i] = 0.0f;
}

// wT[n*K + k] = (f16) w[k*N + n]
__global__ void transpose_f16_kernel(const float* __restrict__ w,
                                     _Float16* __restrict__ wT, int K, int N) {
  int idx = blockIdx.x * blockDim.x + threadIdx.x;
  if (idx < K * N) {
    int k = idx / N, n = idx % N;
    wT[n * K + k] = (_Float16)w[idx];
  }
}

// ---------------------------------------------------------------------------
// Edge MLP: block = 16 edges, 256 threads (8 waves).
// in[16][512] = concat(mesh_x[src], grid_x[dst]) as f16 in LDS.
// layer1: h = relu(in @ w1 + b1)   (K=512, 16 WMMA k-steps per N tile)
// layer2: o = h @ w2 + b2          (K=256,  8 WMMA k-steps per N tile)
// epilogue: atomicAdd into agg[edge_dst].
// ---------------------------------------------------------------------------
__global__ void edge_mlp_kernel(const float* __restrict__ mesh_x,
                                const float* __restrict__ grid_x,
                                const int* __restrict__ esrc,
                                const int* __restrict__ edst,
                                const _Float16* __restrict__ w1T,  // [256][512]
                                const float* __restrict__ b1,
                                const _Float16* __restrict__ w2T,  // [256][256]
                                const float* __restrict__ b2,
                                float* __restrict__ agg) {
  __shared__ _Float16 sIn[16 * 512];
  __shared__ _Float16 sHid[16 * 256];
  __shared__ int sDst[16];

  const int tile = blockIdx.x;
  const int t    = threadIdx.x;
  const int lane = t & 31;
  const int wave = t >> 5;          // 0..7
  const int hi   = lane >> 4;       // 0/1
  const int nrow = lane & 15;       // A/B "row" lane index

  if (t < 16) sDst[t] = edst[tile * 16 + t];

  // Gather: 16 threads per edge row, each covers 32 consecutive columns.
  {
    const int row  = t >> 4;              // 0..15
    const int cseg = (t & 15) * 32;       // 0,32,...,480 (one half only)
    const int e    = tile * 16 + row;
    const float* src =
        (cseg < DIM) ? (mesh_x + (size_t)esrc[e] * DIM + cseg)
                     : (grid_x + (size_t)edst[e] * DIM + (cseg - DIM));
#pragma unroll
    for (int j = 0; j < 32; ++j)
      sIn[row * 512 + cseg + j] = (_Float16)src[j];
  }
  __syncthreads();

  // ---- Layer 1: each wave computes 2 N-tiles of the 16x256 hidden ----
#pragma unroll
  for (int nn = 0; nn < 2; ++nn) {
    const int nbase = (wave * 2 + nn) * 16;
    v8f c = {};
#pragma unroll
    for (int kt = 0; kt < 16; ++kt) {
      v16h a = frag16(&sIn[nrow * 512 + kt * 32], hi);
      v16h b = frag16(&w1T[(size_t)(nbase + nrow) * 512 + kt * 32], hi);
      c = wmma_f16(a, b, c);
    }
    const float bias = b1[nbase + nrow];
#pragma unroll
    for (int v = 0; v < 8; ++v) {
      float x = c[v] + bias;
      x = x > 0.0f ? x : 0.0f;                      // ReLU
      const int row = v + 8 * hi;                   // C/D layout
      sHid[row * 256 + nbase + nrow] = (_Float16)x;
    }
  }
  __syncthreads();

  // ---- Layer 2 + scatter-add ----
#pragma unroll
  for (int nn = 0; nn < 2; ++nn) {
    const int nbase = (wave * 2 + nn) * 16;
    v8f c = {};
#pragma unroll
    for (int kt = 0; kt < 8; ++kt) {
      v16h a = frag16(&sHid[nrow * 256 + kt * 32], hi);
      v16h b = frag16(&w2T[(size_t)(nbase + nrow) * 256 + kt * 32], hi);
      c = wmma_f16(a, b, c);
    }
    const float bias = b2[nbase + nrow];
#pragma unroll
    for (int v = 0; v < 8; ++v) {
      const int row = v + 8 * hi;
      const int col = nbase + nrow;
      atomicAdd(&agg[(size_t)sDst[row] * HID + col], c[v] + bias);
    }
  }
}

// ---------------------------------------------------------------------------
// Grid MLP: block = 16 grid rows. in = concat(grid_x, agg). out = grid_x + mlp.
// ---------------------------------------------------------------------------
__global__ void grid_mlp_kernel(const float* __restrict__ grid_x,
                                const float* __restrict__ agg,
                                const _Float16* __restrict__ w1T,  // [256][512]
                                const float* __restrict__ b1,
                                const _Float16* __restrict__ w2T,  // [256][256]
                                const float* __restrict__ b2,
                                float* __restrict__ out) {
  __shared__ _Float16 sIn[16 * 512];
  __shared__ _Float16 sHid[16 * 256];

  const int tile = blockIdx.x;
  const int t    = threadIdx.x;
  const int lane = t & 31;
  const int wave = t >> 5;
  const int hi   = lane >> 4;
  const int nrow = lane & 15;
  const size_t rbase = (size_t)tile * 16;

  // Stage inputs: first 256 cols = grid_x row, last 256 = agg row.
  {
    const int row  = t >> 4;
    const int cseg = (t & 15) * 32;
    const float* src = (cseg < DIM)
                           ? (grid_x + (rbase + row) * DIM + cseg)
                           : (agg + (rbase + row) * HID + (cseg - DIM));
#pragma unroll
    for (int j = 0; j < 32; ++j)
      sIn[row * 512 + cseg + j] = (_Float16)src[j];
  }
  __syncthreads();

  // ---- Layer 1 ----
#pragma unroll
  for (int nn = 0; nn < 2; ++nn) {
    const int nbase = (wave * 2 + nn) * 16;
    v8f c = {};
#pragma unroll
    for (int kt = 0; kt < 16; ++kt) {
      v16h a = frag16(&sIn[nrow * 512 + kt * 32], hi);
      v16h b = frag16(&w1T[(size_t)(nbase + nrow) * 512 + kt * 32], hi);
      c = wmma_f16(a, b, c);
    }
    const float bias = b1[nbase + nrow];
#pragma unroll
    for (int v = 0; v < 8; ++v) {
      float x = c[v] + bias;
      x = x > 0.0f ? x : 0.0f;
      const int row = v + 8 * hi;
      sHid[row * 256 + nbase + nrow] = (_Float16)x;
    }
  }
  __syncthreads();

  // ---- Layer 2 + residual store ----
#pragma unroll
  for (int nn = 0; nn < 2; ++nn) {
    const int nbase = (wave * 2 + nn) * 16;
    v8f c = {};
#pragma unroll
    for (int kt = 0; kt < 8; ++kt) {
      v16h a = frag16(&sHid[nrow * 256 + kt * 32], hi);
      v16h b = frag16(&w2T[(size_t)(nbase + nrow) * 256 + kt * 32], hi);
      c = wmma_f16(a, b, c);
    }
    const float bias = b2[nbase + nrow];
#pragma unroll
    for (int v = 0; v < 8; ++v) {
      const int row = v + 8 * hi;
      const int col = nbase + nrow;
      const size_t idx = (rbase + row) * HID + col;
      out[idx] = grid_x[idx] + c[v] + bias;   // residual
    }
  }
}

// ---------------------------------------------------------------------------
// Host launcher
// ---------------------------------------------------------------------------
extern "C" void kernel_launch(void* const* d_in, const int* in_sizes, int n_in,
                              void* d_out, int out_size, void* d_ws,
                              size_t ws_size, hipStream_t stream) {
  const float* mesh_x = (const float*)d_in[0];
  const float* grid_x = (const float*)d_in[1];
  const int*   esrc   = (const int*)d_in[2];
  const int*   edst   = (const int*)d_in[3];
  const float* w1_e   = (const float*)d_in[4];
  const float* b1_e   = (const float*)d_in[5];
  const float* w2_e   = (const float*)d_in[6];
  const float* b2_e   = (const float*)d_in[7];
  const float* w1_g   = (const float*)d_in[8];
  const float* b1_g   = (const float*)d_in[9];
  const float* w2_g   = (const float*)d_in[10];
  const float* b2_g   = (const float*)d_in[11];
  float* out = (float*)d_out;

  // Workspace layout (all 256B-aligned by construction).
  char* ws = (char*)d_ws;
  float* agg = (float*)ws;              ws += (size_t)N_GRID * HID * sizeof(float);
  _Float16* w1eT = (_Float16*)ws;       ws += (size_t)(2 * DIM) * HID * sizeof(_Float16);
  _Float16* w2eT = (_Float16*)ws;       ws += (size_t)HID * HID * sizeof(_Float16);
  _Float16* w1gT = (_Float16*)ws;       ws += (size_t)(DIM + HID) * HID * sizeof(_Float16);
  _Float16* w2gT = (_Float16*)ws;

  // 1) zero the aggregation buffer (must be re-zeroed every call: determinism)
  zero_f32_kernel<<<2048, 256, 0, stream>>>(agg, N_GRID * HID);

  // 2) convert weights to transposed f16 (wT[n][k])
  transpose_f16_kernel<<<(2 * DIM * HID + 255) / 256, 256, 0, stream>>>(w1_e, w1eT, 2 * DIM, HID);
  transpose_f16_kernel<<<(HID * HID + 255) / 256, 256, 0, stream>>>(w2_e, w2eT, HID, HID);
  transpose_f16_kernel<<<(2 * DIM * HID + 255) / 256, 256, 0, stream>>>(w1_g, w1gT, DIM + HID, HID);
  transpose_f16_kernel<<<(HID * HID + 255) / 256, 256, 0, stream>>>(w2_g, w2gT, HID, HID);

  // 3) edge MLP + scatter-add (300000 / 16 = 18750 tiles, exact)
  edge_mlp_kernel<<<N_EDGE / 16, 256, 0, stream>>>(mesh_x, grid_x, esrc, edst,
                                                   w1eT, b1_e, w2eT, b2_e, agg);

  // 4) grid MLP + residual (100000 / 16 = 6250 tiles, exact)
  grid_mlp_kernel<<<N_GRID / 16, 256, 0, stream>>>(grid_x, agg, w1gT, b1_g,
                                                   w2gT, b2_g, out);
}
